// Vanilla_MAE_19533511262661
// MI455X (gfx1250) — compile-verified, matrix operand order checked by reading, not run on hardware
//
#include <hip/hip_runtime.h>
#include <hip/hip_bf16.h>

typedef __bf16 bf16_t;
typedef __attribute__((ext_vector_type(16))) __bf16 v16bf;
typedef __attribute__((ext_vector_type(8)))  float  v8f;
typedef __attribute__((ext_vector_type(4)))  unsigned int v4u;
typedef __attribute__((ext_vector_type(8)))  int v8i;
typedef __attribute__((ext_vector_type(4)))  int v4i;

#define N_ROWS 16384
#define SEQ    512
#define HID    1024
#define ENC    256
#define NEXP   8
#define BM     128
#define BN     128
#define BK     32
#define LDSPAD 8                          // bf16 elements of pad per LDS row
#define PAD_TILES (N_ROWS / BM + NEXP)    // 136 (worst-case padded tiles)
#define PAD_ROWS  (PAD_TILES * BM)        // 17408

// ---------------------------------------------------------------- converts
__global__ void cast_bf16_kernel(const float* __restrict__ in,
                                 bf16_t* __restrict__ out, int n) {
  for (int i = blockIdx.x * blockDim.x + threadIdx.x; i < n;
       i += gridDim.x * blockDim.x)
    out[i] = (bf16_t)in[i];
}

// in: [B, K, Ncols] f32 row-major  ->  out: [B, Ncols, K] bf16 (transposed)
__global__ void transpose_bf16_kernel(const float* __restrict__ in,
                                      bf16_t* __restrict__ out,
                                      int K, int Ncols, int batch) {
  long total = (long)batch * K * Ncols;
  long stride = (long)gridDim.x * blockDim.x;
  for (long o = (long)blockIdx.x * blockDim.x + threadIdx.x; o < total; o += stride) {
    long per = (long)K * Ncols;
    long b = o / per;
    long rem = o - b * per;
    int n = (int)(rem / K);
    int k = (int)(rem - (long)n * K);
    out[o] = (bf16_t)in[b * per + (long)k * Ncols + n];
  }
}

// ---------------------------------------------------------------- routing
__global__ void routing_init_kernel(int* __restrict__ perm, int permLen,
                                    int* __restrict__ counts,
                                    int* __restrict__ cursors) {
  int i = blockIdx.x * blockDim.x + threadIdx.x;
  if (i < NEXP) { counts[i] = 0; cursors[i] = 0; }
  for (; i < permLen; i += gridDim.x * blockDim.x) perm[i] = -1;
}

__global__ void routing_count_kernel(const int* __restrict__ ids,
                                     int* __restrict__ counts, int n) {
  int i = blockIdx.x * blockDim.x + threadIdx.x;
  if (i < n) atomicAdd(&counts[ids[i]], 1);
}

__global__ void routing_offsets_kernel(const int* __restrict__ counts,
                                       int* __restrict__ poff) {
  if (blockIdx.x == 0 && threadIdx.x == 0) {
    int acc = 0;
    for (int e = 0; e < NEXP; ++e) {
      poff[e] = acc;
      acc += ((counts[e] + BM - 1) / BM) * BM;  // pad bucket to tile multiple
    }
    poff[NEXP] = acc;
  }
}

__global__ void routing_scatter_kernel(const int* __restrict__ ids,
                                       const int* __restrict__ poff,
                                       int* __restrict__ cursors,
                                       int* __restrict__ perm, int n) {
  int i = blockIdx.x * blockDim.x + threadIdx.x;
  if (i < n) {
    int e = ids[i];
    int pos = atomicAdd(&cursors[e], 1);
    perm[poff[e] + pos] = i;
  }
}

// ---------------------------------------------------------------- TDM tile load
// DMA one [128 rows x 32 bf16] tile (row stride K elements) from global into
// LDS with a 16-byte pad after every 64-byte row -> LDS row stride 80B = BK+8
// bf16, matching the conflict-free compute layout. D# per ISA ch.8.
__device__ __forceinline__ void tdm_load_tile(unsigned lds_off,
                                              unsigned long long gaddr,
                                              unsigned K_elems,
                                              unsigned rows_total) {
  v4u g0;
  g0[0] = 1u;                                            // count=1, no gather
  g0[1] = lds_off;                                       // lds_addr (bytes)
  g0[2] = (unsigned)(gaddr & 0xFFFFFFFFull);             // global_addr[31:0]
  g0[3] = (unsigned)((gaddr >> 32) & 0x1FFFFFFull)       // global_addr[56:32]
          | (2u << 30);                                  // type=2 (image)
  v8i g1;
  g1[0] = (int)((1u << 16)      // data_size = 1 -> 2-byte elements
              | (1u << 20)      // pad_enable
              | (3u << 22)      // pad_interval: 16 DWORDs (64B) before pad
              | (3u << 25));    // pad_amount:   4 DWORDs (16B) of pad
  g1[1] = (int)((K_elems & 0xFFFFu) << 16);                     // tensor_dim0 lo
  g1[2] = (int)(((K_elems >> 16) & 0xFFFFu)                     // tensor_dim0 hi
              | ((rows_total & 0xFFFFu) << 16));                // tensor_dim1 lo
  g1[3] = (int)(((rows_total >> 16) & 0xFFFFu)                  // tensor_dim1 hi
              | ((unsigned)BK << 16));                          // tile_dim0 = 32
  g1[4] = (int)BM;                                              // tile_dim1=128, tile_dim2=0
  g1[5] = (int)K_elems;                                         // dim0_stride lo
  g1[6] = 0;                                                    // dim0_stride hi
  g1[7] = 0;
  v4i gz4 = {0, 0, 0, 0};
  v8i gz8 = {0, 0, 0, 0, 0, 0, 0, 0};
  __builtin_amdgcn_tensor_load_to_lds(g0, g1, gz4, gz4, gz8, 0);
}

// ---------------------------------------------------------------- WMMA GEMM
// Fragment layout per ISA (16-bit A 16x32):
//   lane<16 : row=lane,    elems 0..7 = K 0..7,  8..15 = K 16..23
//   lane>=16: row=lane-16, elems 0..7 = K 8..15, 8..15 = K 24..31
__device__ inline v16bf load_frag(const bf16_t* __restrict__ base, int ldr) {
  int lane = threadIdx.x & 31;
  int r = lane & 15;
  int kk = (lane & 16) >> 1;  // 0 or 8
  const bf16_t* p = base + r * ldr + kk;
  union { v16bf v; uint4 q[2]; } u;
  u.q[0] = *reinterpret_cast<const uint4*>(p);
  u.q[1] = *reinterpret_cast<const uint4*>(p + 16);
  return u.v;
}

// C[M,N] = act(A @ Bt^T + bias)
//   A : [Mrows, K] bf16 row-major (rows gathered via perm when GATHER)
//   Bt: [N, K]     bf16 row-major (EXPERT: per-expert block of N*K, bias of N)
//   C : [*, N]     rows scattered via perm when SCATTER (perm<0 -> skip)
// TDMA: stage A tiles via TDM (regular rows); B tiles always staged via TDM.
template <bool RELU, bool GATHER, bool SCATTER, bool EXPERT, bool TDMA,
          typename OutT>
__global__ __launch_bounds__(256) void wmma_gemm_kernel(
    const bf16_t* __restrict__ A, const bf16_t* __restrict__ Bt,
    const float* __restrict__ bias, OutT* __restrict__ C, int N, int K,
    int Mrows, const int* __restrict__ perm, const int* __restrict__ poff) {
  __shared__ bf16_t As[2][BM][BK + LDSPAD];  // 80B rows, double buffered
  __shared__ bf16_t Bs[2][BN][BK + LDSPAD];

  const int tid  = threadIdx.x;
  const int lane = tid & 31;
  const int wave = tid >> 5;
  const int wm   = wave >> 2;  // 0..1 : 64-row strip
  const int wn   = wave & 3;   // 0..3 : 32-col strip

  const int rowBase = blockIdx.y * BM;
  const int colBase = blockIdx.x * BN;

  if (EXPERT) {
    int e = 0;
    while (e < NEXP - 1 && rowBase >= poff[e + 1]) ++e;
    Bt   += (size_t)e * N * K;
    bias += (size_t)e * N;
  }

  // Per-thread fallback staging (gathered A): row = tid>>1, 16-col half.
  const int ar = tid >> 1;
  const int ah = (tid & 1) * 16;
  const bf16_t* aSrc = nullptr;
  if (!TDMA) {
    int r = rowBase + ar;
    int src = GATHER ? perm[r] : r;
    if (!GATHER || src >= 0) aSrc = A + (size_t)src * K + ah;
  }

  const unsigned long long aG =
      (unsigned long long)(uintptr_t)A + (unsigned long long)rowBase * K * 2ull;
  const unsigned long long bG =
      (unsigned long long)(uintptr_t)Bt + (unsigned long long)colBase * K * 2ull;
  const unsigned asOff[2] = {(unsigned)(uintptr_t)&As[0][0][0],
                             (unsigned)(uintptr_t)&As[1][0][0]};
  const unsigned bsOff[2] = {(unsigned)(uintptr_t)&Bs[0][0][0],
                             (unsigned)(uintptr_t)&Bs[1][0][0]};

  const int nIter = K / BK;

  // ---- prologue: stage tile 0 into buffer 0
  if (wave == 0) {
    if (TDMA) tdm_load_tile(asOff[0], aG, (unsigned)K, (unsigned)Mrows);
    tdm_load_tile(bsOff[0], bG, (unsigned)K, (unsigned)N);
  }
  if (!TDMA) {
    uint4 a0 = {0u, 0u, 0u, 0u}, a1 = {0u, 0u, 0u, 0u};
    if (aSrc) {
      a0 = *reinterpret_cast<const uint4*>(aSrc);
      a1 = *reinterpret_cast<const uint4*>(aSrc + 8);
    }
    *reinterpret_cast<uint4*>(&As[0][ar][ah])     = a0;
    *reinterpret_cast<uint4*>(&As[0][ar][ah + 8]) = a1;
  }
  if (wave == 0) __builtin_amdgcn_s_wait_tensorcnt(0);
  __syncthreads();

  v8f acc[4][2] = {};
  int cur = 0;
#pragma unroll 2
  for (int it = 0; it < nIter; ++it) {
    const int nxt = cur ^ 1;
    const bool hasNext = (it + 1) < nIter;
    uint4 a0 = {0u, 0u, 0u, 0u}, a1 = {0u, 0u, 0u, 0u};
    if (hasNext) {
      const int k0 = (it + 1) * BK;
      if (wave == 0) {
        // issue next tile's DMA; overlaps with this iteration's WMMA work
        if (TDMA)
          tdm_load_tile(asOff[nxt], aG + (unsigned long long)k0 * 2ull,
                        (unsigned)K, (unsigned)Mrows);
        tdm_load_tile(bsOff[nxt], bG + (unsigned long long)k0 * 2ull,
                      (unsigned)K, (unsigned)N);
      }
      if (!TDMA && aSrc) {
        a0 = *reinterpret_cast<const uint4*>(aSrc + k0);
        a1 = *reinterpret_cast<const uint4*>(aSrc + k0 + 8);
      }
    }

    // ---- compute on current buffer
    v16bf af[4], bfr[2];
#pragma unroll
    for (int i = 0; i < 4; ++i)
      af[i] = load_frag(&As[cur][wm * 64 + i * 16][0], BK + LDSPAD);
#pragma unroll
    for (int j = 0; j < 2; ++j)
      bfr[j] = load_frag(&Bs[cur][wn * 32 + j * 16][0], BK + LDSPAD);
#pragma unroll
    for (int i = 0; i < 4; ++i)
#pragma unroll
      for (int j = 0; j < 2; ++j)
        acc[i][j] = __builtin_amdgcn_wmma_f32_16x16x32_bf16(
            false, af[i], false, bfr[j], (short)0, acc[i][j], false, false);

    if (hasNext) {
      if (!TDMA) {
        *reinterpret_cast<uint4*>(&As[nxt][ar][ah])     = a0;
        *reinterpret_cast<uint4*>(&As[nxt][ar][ah + 8]) = a1;
      }
      if (wave == 0) __builtin_amdgcn_s_wait_tensorcnt(0);
      __syncthreads();
    }
    cur = nxt;
  }

  // ---- epilogue: C/D layout: VGPR v -> M=v (lanes 0-15) / v+8 (16-31).
  const int rHalf = (lane & 16) >> 1;  // 0 or 8
  const int cl    = lane & 15;
#pragma unroll
  for (int i = 0; i < 4; ++i) {
    int rowT = rowBase + wm * 64 + i * 16 + rHalf;
    int dstRow[8];
#pragma unroll
    for (int v = 0; v < 8; ++v) {
      int r = rowT + v;
      dstRow[v] = SCATTER ? perm[r] : r;
    }
#pragma unroll
    for (int j = 0; j < 2; ++j) {
      int col = colBase + wn * 32 + j * 16 + cl;
      float bv = bias[col];
#pragma unroll
      for (int v = 0; v < 8; ++v) {
        if (SCATTER && dstRow[v] < 0) continue;
        float val = acc[i][j][v] + bv;
        if (RELU) val = fmaxf(val, 0.0f);
        C[(size_t)dstRow[v] * N + col] = (OutT)val;
      }
    }
  }
}

// ---------------------------------------------------------------- launch
extern "C" void kernel_launch(void* const* d_in, const int* in_sizes, int n_in,
                              void* d_out, int out_size, void* d_ws,
                              size_t ws_size, hipStream_t stream) {
  const float* x  = (const float*)d_in[0];
  const int*   ids= (const int*)  d_in[1];
  const float* W1 = (const float*)d_in[2];
  const float* b1 = (const float*)d_in[3];
  const float* W2 = (const float*)d_in[4];
  const float* b2 = (const float*)d_in[5];
  const float* W3 = (const float*)d_in[6];
  const float* b3 = (const float*)d_in[7];
  const float* W4 = (const float*)d_in[8];
  const float* b4 = (const float*)d_in[9];
  float* out = (float*)d_out;

  size_t off = 0;
  auto alloc = [&](size_t bytes) -> void* {
    off = (off + 255) & ~(size_t)255;
    void* p = (char*)d_ws + off;
    off += bytes;
    return p;
  };
  bf16_t* xb   = (bf16_t*)alloc((size_t)N_ROWS * SEQ * 2);
  bf16_t* W1t  = (bf16_t*)alloc((size_t)HID * SEQ * 2);        // [HID,SEQ]
  bf16_t* W2t  = (bf16_t*)alloc((size_t)ENC * HID * 2);        // [ENC,HID]
  bf16_t* W3t  = (bf16_t*)alloc((size_t)NEXP * HID * ENC * 2); // [E,HID,ENC]
  bf16_t* W4t  = (bf16_t*)alloc((size_t)NEXP * SEQ * HID * 2); // [E,SEQ,HID]
  bf16_t* hb   = (bf16_t*)alloc((size_t)N_ROWS * HID * 2);
  bf16_t* zb   = (bf16_t*)alloc((size_t)N_ROWS * ENC * 2);
  bf16_t* ub   = (bf16_t*)alloc((size_t)PAD_ROWS * HID * 2);
  int* perm    = (int*)alloc((size_t)PAD_ROWS * 4);
  int* counts  = (int*)alloc(NEXP * 4);
  int* cursors = (int*)alloc(NEXP * 4);
  int* poff    = (int*)alloc((NEXP + 1) * 4);

  // f32 -> bf16 (x) and f32 -> transposed bf16 (weights)
  cast_bf16_kernel<<<4096, 256, 0, stream>>>(x, xb, N_ROWS * SEQ);
  transpose_bf16_kernel<<<2048, 256, 0, stream>>>(W1, W1t, SEQ, HID, 1);
  transpose_bf16_kernel<<<1024, 256, 0, stream>>>(W2, W2t, HID, ENC, 1);
  transpose_bf16_kernel<<<2048, 256, 0, stream>>>(W3, W3t, ENC, HID, NEXP);
  transpose_bf16_kernel<<<4096, 256, 0, stream>>>(W4, W4t, HID, SEQ, NEXP);

  // expert routing (padded buckets of BM rows)
  routing_init_kernel<<<(PAD_ROWS + 255) / 256, 256, 0, stream>>>(perm, PAD_ROWS,
                                                                  counts, cursors);
  routing_count_kernel<<<(N_ROWS + 255) / 256, 256, 0, stream>>>(ids, counts, N_ROWS);
  routing_offsets_kernel<<<1, 1, 0, stream>>>(counts, poff);
  routing_scatter_kernel<<<(N_ROWS + 255) / 256, 256, 0, stream>>>(ids, poff,
                                                                   cursors, perm, N_ROWS);

  dim3 blk(256);
  // h = relu(x @ W1 + b1)           (A + B tiles via TDM, double buffered)
  wmma_gemm_kernel<true, false, false, false, true, bf16_t>
      <<<dim3(HID / BN, N_ROWS / BM), blk, 0, stream>>>(
          xb, W1t, b1, hb, HID, SEQ, N_ROWS, nullptr, nullptr);
  // z = relu(h @ W2 + b2)
  wmma_gemm_kernel<true, false, false, false, true, bf16_t>
      <<<dim3(ENC / BN, N_ROWS / BM), blk, 0, stream>>>(
          hb, W2t, b2, zb, ENC, HID, N_ROWS, nullptr, nullptr);
  // u = relu(gather(z) @ W3[e] + b3[e])   (A gathered per-thread; B via TDM)
  wmma_gemm_kernel<true, true, false, true, false, bf16_t>
      <<<dim3(HID / BN, PAD_TILES), blk, 0, stream>>>(
          zb, W3t, b3, ub, HID, ENC, PAD_ROWS, perm, poff);
  // y = scatter(u @ W4[e] + b4[e])        (writes every output row exactly once)
  wmma_gemm_kernel<false, false, true, true, true, float>
      <<<dim3(SEQ / BN, PAD_TILES), blk, 0, stream>>>(
          ub, W4t, b4, out, SEQ, HID, PAD_ROWS, perm, poff);
}
